// ReadoutLayer_34213709480841
// MI455X (gfx1250) — matrix operand order repaired and checked by reading, not compiled, not run on hardware
//
#include <hip/hip_runtime.h>
#include <hip/hip_bf16.h>

typedef __attribute__((ext_vector_type(16))) __bf16 v16bf;
typedef __attribute__((ext_vector_type(8)))  float  v8f;

#define ALPHA_LO 0.8187307530779818f
#define ALPHA_HI 0.9607894391523232f

constexpr int Bsz    = 64;
constexpr int T      = 2000;
constexpr int I      = 512;
constexpr int H      = 512;
constexpr int TCHUNK = 16;
constexpr int NCHUNK = T / TCHUNK;   // 125
constexpr int NWAVES = 16;
constexpr int BLOCK  = NWAVES * 32;  // 512 threads = 16 wave32

__device__ __forceinline__ unsigned short f2bf(float f) {
    unsigned int u = __float_as_uint(f);
    // round-to-nearest-even bf16
    return (unsigned short)((u + 0x7FFFu + ((u >> 16) & 1u)) >> 16);
}

// ---- Kernel 1: W fp32 [H,I] -> bf16 row-major in workspace ----
__global__ void convW_kernel(const float* __restrict__ W,
                             unsigned short* __restrict__ Wb) {
    int i = blockIdx.x * blockDim.x + threadIdx.x;
    if (i < H * I) Wb[i] = f2bf(W[i]);
}

// ---- Kernel 2: fused GEMM (WMMA bf16) + EMA scan + softmax accumulation ----
__global__ __launch_bounds__(BLOCK)
void readout_fused_kernel(const float* __restrict__ x,
                          const unsigned short* __restrict__ Wb,
                          const float* __restrict__ alpha,
                          const float* __restrict__ u0,
                          float* __restrict__ out) {
    __shared__ unsigned short sA[TCHUNK][I];  // 16 KB: x chunk, bf16
    __shared__ float          swx[TCHUNK][H]; // 32 KB: Wx chunk, f32
    __shared__ float          su[2][H];       //  4 KB: u state, double-buffered

    const int b    = blockIdx.x;
    const int tid  = threadIdx.x;
    const int wave = tid >> 5;
    const int lane = tid & 31;
    const int half = lane >> 4;   // lane-half selects K sub-block (ISA 7.12.2)
    const int l15  = lane & 15;

    // Per-lane alpha/beta/out registers: lane owns h = lane + 32*r, r=0..15
    float a_r[16], b_r[16], out_acc[16];
#pragma unroll
    for (int r = 0; r < 16; ++r) {
        float a = alpha[lane + 32 * r];
        a = fminf(fmaxf(a, ALPHA_LO), ALPHA_HI);
        a_r[r] = a;
        b_r[r] = 1.0f - a;
        out_acc[r] = 0.0f;
    }
    for (int h = tid; h < H; h += BLOCK) su[0][h] = u0[(size_t)b * H + h];
    __syncthreads();

    const float* xb = x + (size_t)b * T * I;
    unsigned short* sAf = &sA[0][0];

    // ---- Software pipeline: prefetch chunk 0 of x into registers ----
    float4 xreg[4];
#pragma unroll
    for (int q = 0; q < 4; ++q)
        xreg[q] = *(const float4*)(xb + tid * 16 + q * 4);

    for (int c = 0; c < NCHUNK; ++c) {
        // ---- Phase 1: convert prefetched x chunk -> bf16 LDS ----
        {
            const int base = tid * 16;           // 8192 elems / 512 threads
#pragma unroll
            for (int q = 0; q < 4; ++q) {
                float4 v = xreg[q];
                unsigned int p0 = (unsigned int)f2bf(v.x) |
                                  ((unsigned int)f2bf(v.y) << 16);
                unsigned int p1 = (unsigned int)f2bf(v.z) |
                                  ((unsigned int)f2bf(v.w) << 16);
                *(uint2*)(sAf + base + q * 4) = make_uint2(p0, p1);
            }
        }
        __syncthreads();

        // Issue global loads for the NEXT chunk now; they complete in the
        // shadow of the WMMA + scan phases below.
        if (c + 1 < NCHUNK) {
            const float* src = xb + (size_t)(c + 1) * TCHUNK * I + tid * 16;
#pragma unroll
            for (int q = 0; q < 4; ++q)
                xreg[q] = *(const float4*)(src + q * 4);
        }

        // Opaque zero offset, re-materialized per chunk iteration.  Keeps the
        // base pointer (Wb) visible so addrspace inference still yields
        // global_load (not flat_load coupled to DScnt), while preventing LICM
        // from hoisting the 32 B-fragment loads out of the chunk loop (which
        // previously caused 256-VGPR pressure + scratch spills).
        unsigned wofs = 0;
        asm volatile("" : "+v"(wofs));

        // ---- Phase 2: WMMA — wave owns N-tiles {2*wave, 2*wave+1} ----
#pragma unroll
        for (int nt = 0; nt < 2; ++nt) {
            const int n0 = (2 * wave + nt) * 16;
            // Two accumulators (even/odd K-step) break the WMMA->WMMA RAW
            // chain on the C/D operand (ISA 7.12.1 hazard: 5 slots for bf16).
            v8f acc0 = {0.f, 0.f, 0.f, 0.f, 0.f, 0.f, 0.f, 0.f};
            v8f acc1 = {0.f, 0.f, 0.f, 0.f, 0.f, 0.f, 0.f, 0.f};
            const unsigned short* arow = &sA[l15][0];
            const unsigned short* wrow =
                Wb + (size_t)(n0 + l15) * I + half * 16 + wofs;
#pragma unroll
            for (int k0 = 0; k0 < I; k0 += 64) {
                union { uint4 q[2]; v16bf v; } af0, af1, bf0, bf1;
                // A fragments (16x32 bf16): lane-half 0 -> K {0..15},
                // half 1 -> K {16..31}, row = l15
                af0.q[0] = *(const uint4*)(arow + k0 + half * 16);
                af0.q[1] = *(const uint4*)(arow + k0 + 8 + half * 16);
                af1.q[0] = *(const uint4*)(arow + k0 + 32 + half * 16);
                af1.q[1] = *(const uint4*)(arow + k0 + 40 + half * 16);
                // B fragments (32x16 bf16): lane = column h = n0+l15,
                // contiguous K from row-major W (K = i)
                bf0.q[0] = *(const uint4*)(wrow + k0);
                bf0.q[1] = *(const uint4*)(wrow + k0 + 8);
                bf1.q[0] = *(const uint4*)(wrow + k0 + 32);
                bf1.q[1] = *(const uint4*)(wrow + k0 + 40);
                acc0 = __builtin_amdgcn_wmma_f32_16x16x32_bf16(
                    false, af0.v, false, bf0.v, (short)0, acc0, false, false);
                acc1 = __builtin_amdgcn_wmma_f32_16x16x32_bf16(
                    false, af1.v, false, bf1.v, (short)0, acc1, false, false);
            }
            // D layout: VGPR v, lanes 0-15 -> M=v, lanes 16-31 -> M=v+8
            const int mbase = half * 8;
#pragma unroll
            for (int v = 0; v < 8; ++v)
                swx[mbase + v][n0 + l15] = acc0[v] + acc1[v];
        }
        __syncthreads();

        // ---- Phase 3: wave w handles timestep t0+w (linear recurrence) ----
        const int rd = c & 1, wr = rd ^ 1;
        float uloc[16];
#pragma unroll
        for (int r = 0; r < 16; ++r) {
            const int h = lane + 32 * r;
            float u = su[rd][h];
            for (int j = 0; j <= wave; ++j)
                u = a_r[r] * u + b_r[r] * swx[j][h];
            uloc[r] = u;
        }
        if (wave == NWAVES - 1) {
#pragma unroll
            for (int r = 0; r < 16; ++r) su[wr][lane + 32 * r] = uloc[r];
        }
        // Intra-wave softmax over H=512 (16 values/lane x 32 lanes)
        float m = -__builtin_inff();
#pragma unroll
        for (int r = 0; r < 16; ++r) m = fmaxf(m, uloc[r]);
        for (int off = 16; off >= 1; off >>= 1)
            m = fmaxf(m, __shfl_xor(m, off));
        float e[16], s = 0.0f;
#pragma unroll
        for (int r = 0; r < 16; ++r) { e[r] = __expf(uloc[r] - m); s += e[r]; }
        for (int off = 16; off >= 1; off >>= 1)
            s += __shfl_xor(s, off);
        const float inv = 1.0f / s;
#pragma unroll
        for (int r = 0; r < 16; ++r) out_acc[r] += e[r] * inv;
        __syncthreads();  // swx + su[rd] consumed; safe to overwrite next chunk
    }

    // ---- Epilogue: reduce 16 per-wave accumulators (reuse swx) ----
#pragma unroll
    for (int r = 0; r < 16; ++r) swx[wave][lane + 32 * r] = out_acc[r];
    __syncthreads();
    float acc = 0.0f;
#pragma unroll
    for (int w = 0; w < NWAVES; ++w) acc += swx[w][tid];
    out[(size_t)b * H + tid] = acc;
}

extern "C" void kernel_launch(void* const* d_in, const int* in_sizes, int n_in,
                              void* d_out, int out_size, void* d_ws, size_t ws_size,
                              hipStream_t stream) {
    (void)in_sizes; (void)n_in; (void)out_size; (void)ws_size;
    const float* x     = (const float*)d_in[0];
    const float* W     = (const float*)d_in[1];
    const float* alpha = (const float*)d_in[2];
    const float* u0    = (const float*)d_in[3];
    unsigned short* Wb = (unsigned short*)d_ws;   // 512 KB bf16 copy of W

    convW_kernel<<<(H * I + 255) / 256, 256, 0, stream>>>(W, Wb);
    readout_fused_kernel<<<Bsz, BLOCK, 0, stream>>>(x, Wb, alpha, u0,
                                                    (float*)d_out);
}
